// Transformer_2817498546419
// MI455X (gfx1250) — matrix operand order repaired and checked by reading, not compiled, not run on hardware
//
#include <hip/hip_runtime.h>
#include <math.h>
#include <stdint.h>

// ---------------------------------------------------------------------------
// Transformer encoder-decoder for MI455X (gfx1250), compile-only tuning.
// GEMMs: v_wmma_f32_16x16x32_f16, f16 operands (pre-converted once), fp32 acc.
// B tiles staged in LDS via CDNA5 async global->LDS DMA (ASYNCcnt-tracked),
// shared by all 8 waves of the block; A fragments stream from L2 (f16).
// ---------------------------------------------------------------------------

typedef _Float16 f16;
typedef __attribute__((ext_vector_type(16))) _Float16 v16h;
typedef __attribute__((ext_vector_type(8)))  _Float16 v8h;
typedef __attribute__((ext_vector_type(8)))  float    v8f;

#define D_MODEL 512
#define NHEAD   8
#define DKH     64
#define NLAYER  6
#define DFF_    2048
#define BATCH   4
#define SEQ     512
#define MTOK    (BATCH*SEQ)
#define VTGT    32000
#define PADID   1
#define LNEPS   1e-5f

#define LDS_PITCH 80   // 64B of K-halves per row + 16B pad (conflict-free reads)

// ---------------- CDNA5 async global->LDS helpers ----------------
__device__ inline void async_load_b128(uint32_t lds_off, const void* gaddr) {
  asm volatile("global_load_async_to_lds_b128 %0, %1, off"
               :: "v"(lds_off), "v"((uint64_t)(uintptr_t)gaddr)
               : "memory");
}
__device__ inline void wait_asynccnt0() {
  asm volatile("s_wait_asynccnt 0" ::: "memory");
}

// ---------------- wave reductions (wave32) ----------------
__device__ inline float wsum(float v) {
  for (int o = 16; o > 0; o >>= 1) v += __shfl_xor(v, o, 32);
  return v;
}
__device__ inline float wmax(float v) {
  for (int o = 16; o > 0; o >>= 1) v = fmaxf(v, __shfl_xor(v, o, 32));
  return v;
}

// ---------------- fp32 -> f16 conversion ----------------
__global__ void cvt_kernel(f16* __restrict__ o, const float* __restrict__ i, int n) {
  int idx = blockIdx.x * blockDim.x + threadIdx.x;
  if (idx < n) o[idx] = (f16)i[idx];
}

// ---------------- embedding + positional encoding (fp32 residual) ----------
__global__ void embed_kernel(float* __restrict__ out, const int* __restrict__ tok,
                             const float* __restrict__ emb) {
  int idx = blockIdx.x * blockDim.x + threadIdx.x;          // B*SEQ*D total
  int d = idx % D_MODEL;
  int s = (idx / D_MODEL) % SEQ;
  int b = idx / (D_MODEL * SEQ);
  int t = tok[b * SEQ + s];
  float e = (float)(d & ~1);
  float angle = (float)s * __expf(-e * (logf(10000.f) / (float)D_MODEL));
  float pe = (d & 1) ? __cosf(angle) : __sinf(angle);
  out[idx] = emb[(size_t)t * D_MODEL + d] * 22.62741699796952f /*sqrt(512)*/ + pe;
}

// ---------------- layernorm: fp32 in -> f16 out, one wave per row ----------
__global__ void layernorm_kernel(f16* __restrict__ out, const float* __restrict__ in,
                                 const float* __restrict__ g, const float* __restrict__ bta) {
  int row  = (blockIdx.x * blockDim.x + threadIdx.x) >> 5;
  int lane = threadIdx.x & 31;
  const float* x = in + (size_t)row * D_MODEL;
  float vals[16];
  float s = 0.f;
  #pragma unroll
  for (int j = 0; j < 16; ++j) { vals[j] = x[lane + j * 32]; s += vals[j]; }
  float mean = wsum(s) * (1.f / D_MODEL);
  float v = 0.f;
  #pragma unroll
  for (int j = 0; j < 16; ++j) { float d0 = vals[j] - mean; v += d0 * d0; }
  float inv = rsqrtf(wsum(v) * (1.f / D_MODEL) + LNEPS);
  f16* o = out + (size_t)row * D_MODEL;
  #pragma unroll
  for (int j = 0; j < 16; ++j) {
    int d = lane + j * 32;
    o[d] = (f16)((vals[j] - mean) * inv * g[d] + bta[d]);
  }
}

// ---------------- V -> V^T  (f16 [B,S,D] -> f16 [B,D,S]) ----------------
__global__ void vtrans_kernel(unsigned short* __restrict__ out,
                              const unsigned short* __restrict__ in) {
  int idx = blockIdx.x * blockDim.x + threadIdx.x;          // B*SEQ*D total
  int d = idx % D_MODEL;
  int s = (idx / D_MODEL) % SEQ;
  int b = idx / (D_MODEL * SEQ);
  out[(size_t)b * D_MODEL * SEQ + (size_t)d * SEQ + s] = in[idx];
}

// ---------------- masked softmax: fp32 scores -> f16 probs ----------------
__global__ void softmax_mask_kernel(const float* __restrict__ sc, f16* __restrict__ pr,
                                    const int* __restrict__ tok, int causal) {
  int row  = (blockIdx.x * blockDim.x + threadIdx.x) >> 5;  // B*H*SEQ rows
  int lane = threadIdx.x & 31;
  int q  = row % SEQ;
  int b  = row / (SEQ * NHEAD);
  const float* r = sc + (size_t)row * SEQ;
  f16* p = pr + (size_t)row * SEQ;
  const int* tk = tok + (size_t)b * SEQ;
  const float NEGINF = -__builtin_inff();
  float vals[16];
  float mx = NEGINF;
  #pragma unroll
  for (int j = 0; j < 16; ++j) {
    int k = lane + j * 32;
    bool masked = (tk[k] == PADID) || (causal && (k > q));
    float v = masked ? NEGINF : r[k];
    vals[j] = v;
    mx = fmaxf(mx, v);
  }
  mx = wmax(mx);
  if (mx == NEGINF) {                      // fully masked row -> nan_to_num -> 0
    #pragma unroll
    for (int j = 0; j < 16; ++j) p[lane + j * 32] = (f16)0.f;
    return;
  }
  float sum = 0.f;
  #pragma unroll
  for (int j = 0; j < 16; ++j) {
    float e = (vals[j] == NEGINF) ? 0.f : __expf(vals[j] - mx);
    vals[j] = e;
    sum += e;
  }
  float inv = 1.f / wsum(sum);
  #pragma unroll
  for (int j = 0; j < 16; ++j) p[lane + j * 32] = (f16)(vals[j] * inv);
}

// ---------------- WMMA GEMM ------------------------------------------------
// C/C16 = scale*(A @ B^T) + bias + R   (A:[M,K] f16, B:[N,K] f16)
// Block: 256 thr = 8 waves, tile 128(M) x 64(N); each wave owns 16x64.
// B tile (64x32 f16) staged in LDS once per K-step via async DMA, shared by
// all waves.  Batched via blockIdx.z with (z/Hsplit, z%Hsplit) offsets.
__global__ __launch_bounds__(256) void gemm_wmma_kernel(
    float* C, f16* C16, const f16* A, const f16* Bw,
    const float* __restrict__ bias, const float* R,
    int M, int N, int K, int lda, int ldb, int ldc,
    long long a_ob, long long a_oh, long long b_ob, long long b_oh,
    long long c_ob, long long c_oh, int Hsplit,
    float scale, int relu)
{
  __shared__ char ltile[64 * LDS_PITCH];

  int z  = blockIdx.z;
  int bi = z / Hsplit, hi = z % Hsplit;
  A  += bi * a_ob + hi * a_oh;
  Bw += bi * b_ob + hi * b_oh;
  if (C)   C   += bi * c_ob + hi * c_oh;
  if (C16) C16 += bi * c_ob + hi * c_oh;
  if (R)   R   += bi * c_ob + hi * c_oh;

  int tid   = threadIdx.x;
  int lane  = tid & 31;
  int wave  = tid >> 5;
  int mtile = blockIdx.y * 128 + wave * 16;
  int ntile = blockIdx.x * 64;
  int l15   = lane & 15;
  int h16   = lane >> 4;      // lane-half selector

  // staging: thread t moves one 16B chunk: row n = t>>2, quad q = t&3
  int srow = tid >> 2, squad = tid & 3;
  const f16* sgp  = Bw + (size_t)(ntile + srow) * ldb + squad * 8;
  uint32_t  soff  = (uint32_t)(uintptr_t)(ltile + srow * LDS_PITCH + squad * 16);

  v8f acc[4] = {};
  const f16* arow = A + (size_t)(mtile + l15) * lda + h16 * 8;

  for (int kk = 0; kk < K; kk += 32) {
    async_load_b128(soff, sgp + kk);          // 256 lanes x 16B = 4KB B tile
    wait_asynccnt0();
    __syncthreads();

    // A fragment (16x32): lane = M; halves 0-7 <- K[kk+8*h16..), 8-15 <- +16
    union { v16h v; v8h h[2]; } af;
    af.h[0] = *(const v8h*)(arow + kk);
    af.h[1] = *(const v8h*)(arow + kk + 16);

    #pragma unroll
    for (int nt = 0; nt < 4; ++nt) {
      // B fragment (32x16) from LDS: lane = N; halves 0-7 <- K[8*h16..), 8-15 <- +16
      const char* lp = ltile + (nt * 16 + l15) * LDS_PITCH + h16 * 16;
      union { v16h v; v8h h[2]; } bf;
      bf.h[0] = *(const v8h*)(lp);
      bf.h[1] = *(const v8h*)(lp + 32);
      acc[nt] = __builtin_amdgcn_wmma_f32_16x16x32_f16(
          false, af.v, false, bf.v, (short)0, acc[nt], false, false);
    }
    __syncthreads();                          // done reading before next stage
  }

  // Epilogue: C/D layout -> vgpr r: lanes0-15 M=r, lanes16-31 M=8+r; N=lane&15
  #pragma unroll
  for (int nt = 0; nt < 4; ++nt) {
    int col = ntile + nt * 16 + l15;
    float bz = bias ? bias[col] : 0.f;
    #pragma unroll
    for (int r = 0; r < 8; ++r) {
      int row = mtile + r + h16 * 8;
      float v = acc[nt][r] * scale + bz;
      if (relu) v = fmaxf(v, 0.f);
      size_t off = (size_t)row * ldc + col;
      if (R)   v += R[off];
      if (C)   C[off]   = v;
      if (C16) C16[off] = (f16)v;
    }
  }
}

// ---------------------------------------------------------------------------
// Host-side orchestration
// ---------------------------------------------------------------------------
static void gemm(hipStream_t st, float* C, f16* C16, const f16* A, const f16* Bw,
                 const float* bias, const float* R,
                 int M, int N, int K, int lda, int ldb, int ldc,
                 long long a_ob, long long a_oh, long long b_ob, long long b_oh,
                 long long c_ob, long long c_oh, int Z, int Hs,
                 float scale, int relu)
{
  dim3 g(N / 64, M / 128, Z), b(256);
  gemm_wmma_kernel<<<g, b, 0, st>>>(C, C16, A, Bw, bias, R, M, N, K, lda, ldb, ldc,
                                    a_ob, a_oh, b_ob, b_oh, c_ob, c_oh, Hs,
                                    scale, relu);
}

static void cvt(hipStream_t st, f16* dst, const float* src, int n) {
  cvt_kernel<<<(n + 255) / 256, 256, 0, st>>>(dst, src, n);
}

static void layernorm(hipStream_t st, f16* out, const float* in,
                      const float* g, const float* b) {
  layernorm_kernel<<<MTOK / 8, 256, 0, st>>>(out, in, g, b);   // 8 waves/block
}

// self/cross attention; all sequence lengths are 512 here.
static void attention(hipStream_t st,
                      float* xio,                       // fp32 residual in/out
                      const f16* qsrc, const f16* kvsrc,
                      const float* Wq, const float* Wk, const float* Wv, const float* Wo,
                      const int* mask_tok, int causal,
                      f16* wb,                          // f16 weight slab (>= 4*D*D)
                      f16* q16, f16* k16, f16* v16, f16* vt16, f16* att16,
                      float* sc, f16* p16)
{
  const int D = D_MODEL, S = SEQ;
  const int DD = D * D;
  cvt(st, wb,          Wq, DD);
  cvt(st, wb + DD,     Wk, DD);
  cvt(st, wb + 2 * DD, Wv, DD);
  cvt(st, wb + 3 * DD, Wo, DD);
  gemm(st, nullptr, q16, qsrc,  wb,          nullptr, nullptr, MTOK, D, D, D, D, D,
       0,0,0,0,0,0, 1,1, 1.f, 0);
  gemm(st, nullptr, k16, kvsrc, wb + DD,     nullptr, nullptr, MTOK, D, D, D, D, D,
       0,0,0,0,0,0, 1,1, 1.f, 0);
  gemm(st, nullptr, v16, kvsrc, wb + 2 * DD, nullptr, nullptr, MTOK, D, D, D, D, D,
       0,0,0,0,0,0, 1,1, 1.f, 0);
  vtrans_kernel<<<(MTOK * D) / 256, 256, 0, st>>>((unsigned short*)vt16,
                                                  (const unsigned short*)v16);
  // scores[b,h,q,k] = (Q K^T)/sqrt(dk)  (fp32)
  gemm(st, sc, nullptr, q16, k16, nullptr, nullptr, S, S, DKH, D, D, S,
       (long long)S * D, DKH, (long long)S * D, DKH,
       (long long)NHEAD * S * S, (long long)S * S,
       BATCH * NHEAD, NHEAD, 0.125f, 0);
  softmax_mask_kernel<<<(BATCH * NHEAD * S) / 8, 256, 0, st>>>(sc, p16, mask_tok, causal);
  // att[b,q,h*dk+i] = P @ V   (f16 out)
  gemm(st, nullptr, att16, p16, vt16, nullptr, nullptr, S, DKH, S, S, S, D,
       (long long)NHEAD * S * S, (long long)S * S,
       (long long)D * S, (long long)DKH * S,
       (long long)S * D, DKH,
       BATCH * NHEAD, NHEAD, 1.f, 0);
  // xio = xio + att @ Wo^T  (fp32 residual)
  gemm(st, xio, nullptr, att16, wb + 3 * DD, nullptr, xio, MTOK, D, D, D, D, D,
       0,0,0,0,0,0, 1,1, 1.f, 0);
}

extern "C" void kernel_launch(void* const* d_in, const int* in_sizes, int n_in,
                              void* d_out, int out_size, void* d_ws, size_t ws_size,
                              hipStream_t stream) {
  const int*   src       = (const int*)  d_in[0];
  const int*   tgt       = (const int*)  d_in[1];
  const float* src_embed = (const float*)d_in[2];
  const float* tgt_embed = (const float*)d_in[3];
  const float* enc_Wq = (const float*)d_in[4],  *enc_Wk = (const float*)d_in[5];
  const float* enc_Wv = (const float*)d_in[6],  *enc_Wo = (const float*)d_in[7];
  const float* enc_w1 = (const float*)d_in[8],  *enc_b1 = (const float*)d_in[9];
  const float* enc_w2 = (const float*)d_in[10], *enc_b2 = (const float*)d_in[11];
  const float* enc_ln1g = (const float*)d_in[12], *enc_ln1b = (const float*)d_in[13];
  const float* enc_ln2g = (const float*)d_in[14], *enc_ln2b = (const float*)d_in[15];
  const float* enc_lnfg = (const float*)d_in[16], *enc_lnfb = (const float*)d_in[17];
  const float* dec_sWq = (const float*)d_in[18], *dec_sWk = (const float*)d_in[19];
  const float* dec_sWv = (const float*)d_in[20], *dec_sWo = (const float*)d_in[21];
  const float* dec_cWq = (const float*)d_in[22], *dec_cWk = (const float*)d_in[23];
  const float* dec_cWv = (const float*)d_in[24], *dec_cWo = (const float*)d_in[25];
  const float* dec_w1 = (const float*)d_in[26], *dec_b1 = (const float*)d_in[27];
  const float* dec_w2 = (const float*)d_in[28], *dec_b2 = (const float*)d_in[29];
  const float* dec_ln1g = (const float*)d_in[30], *dec_ln1b = (const float*)d_in[31];
  const float* dec_ln2g = (const float*)d_in[32], *dec_ln2b = (const float*)d_in[33];
  const float* dec_ln3g = (const float*)d_in[34], *dec_ln3b = (const float*)d_in[35];
  const float* dec_lnfg = (const float*)d_in[36], *dec_lnfb = (const float*)d_in[37];
  const float* out_W = (const float*)d_in[38], *out_b = (const float*)d_in[39];
  float* logits = (float*)d_out;

  // workspace carve-up
  char* wp = (char*)d_ws;
  auto take = [&](size_t bytes) { char* p = wp; wp += (bytes + 255) & ~(size_t)255; return p; };
  const size_t XN = (size_t)MTOK * D_MODEL;
  float* xe    = (float*)take(XN * 4);                         // enc residual
  float* y     = (float*)take(XN * 4);                         // dec residual
  float* sc    = (float*)take((size_t)BATCH * NHEAD * SEQ * SEQ * 4);
  f16* nrm16   = (f16*)take(XN * 2);
  f16* mem16   = (f16*)take(XN * 2);
  f16* q16     = (f16*)take(XN * 2);
  f16* k16     = (f16*)take(XN * 2);
  f16* v16     = (f16*)take(XN * 2);
  f16* vt16    = (f16*)take(XN * 2);
  f16* att16   = (f16*)take(XN * 2);
  f16* h1_16   = (f16*)take((size_t)MTOK * DFF_ * 2);
  f16* p16     = (f16*)take((size_t)BATCH * NHEAD * SEQ * SEQ * 2);
  f16* wb      = (f16*)take((size_t)VTGT * D_MODEL * 2);       // f16 weight slab

  const int D = D_MODEL;
  const long long DD = (long long)D * D, FD = (long long)DFF_ * D;

  // ---------------- Encoder ----------------
  embed_kernel<<<(MTOK * D) / 256, 256, 0, stream>>>(xe, src, src_embed);
  for (int l = 0; l < NLAYER; ++l) {
    layernorm(stream, nrm16, xe, enc_ln1g + l * D, enc_ln1b + l * D);
    attention(stream, xe, nrm16, nrm16,
              enc_Wq + l * DD, enc_Wk + l * DD, enc_Wv + l * DD, enc_Wo + l * DD,
              src, 0, wb, q16, k16, v16, vt16, att16, sc, p16);
    layernorm(stream, nrm16, xe, enc_ln2g + l * D, enc_ln2b + l * D);
    cvt(stream, wb,      enc_w1 + l * FD, (int)FD);
    cvt(stream, wb + FD, enc_w2 + l * FD, (int)FD);
    gemm(stream, nullptr, h1_16, nrm16, wb, enc_b1 + l * DFF_, nullptr,
         MTOK, DFF_, D, D, D, DFF_, 0,0,0,0,0,0, 1,1, 1.f, 1);
    gemm(stream, xe, nullptr, h1_16, wb + FD, enc_b2 + l * D, xe,
         MTOK, D, DFF_, DFF_, DFF_, D, 0,0,0,0,0,0, 1,1, 1.f, 0);
  }
  layernorm(stream, mem16, xe, enc_lnfg, enc_lnfb);

  // ---------------- Decoder ----------------
  embed_kernel<<<(MTOK * D) / 256, 256, 0, stream>>>(y, tgt, tgt_embed);
  for (int l = 0; l < NLAYER; ++l) {
    layernorm(stream, nrm16, y, dec_ln1g + l * D, dec_ln1b + l * D);
    attention(stream, y, nrm16, nrm16,
              dec_sWq + l * DD, dec_sWk + l * DD, dec_sWv + l * DD, dec_sWo + l * DD,
              tgt, 1, wb, q16, k16, v16, vt16, att16, sc, p16);
    layernorm(stream, nrm16, y, dec_ln2g + l * D, dec_ln2b + l * D);
    attention(stream, y, nrm16, mem16,
              dec_cWq + l * DD, dec_cWk + l * DD, dec_cWv + l * DD, dec_cWo + l * DD,
              src, 0, wb, q16, k16, v16, vt16, att16, sc, p16);
    layernorm(stream, nrm16, y, dec_ln3g + l * D, dec_ln3b + l * D);
    cvt(stream, wb,      dec_w1 + l * FD, (int)FD);
    cvt(stream, wb + FD, dec_w2 + l * FD, (int)FD);
    gemm(stream, nullptr, h1_16, nrm16, wb, dec_b1 + l * DFF_, nullptr,
         MTOK, DFF_, D, D, D, DFF_, 0,0,0,0,0,0, 1,1, 1.f, 1);
    gemm(stream, y, nullptr, h1_16, wb + FD, dec_b2 + l * D, y,
         MTOK, D, DFF_, DFF_, DFF_, D, 0,0,0,0,0,0, 1,1, 1.f, 0);
  }
  layernorm(stream, nrm16, y, dec_lnfg, dec_lnfb);

  // ---------------- Output projection: [2048,512] @ [32000,512]^T -----------
  cvt(stream, wb, out_W, VTGT * D);
  gemm(stream, logits, nullptr, nrm16, wb, out_b, nullptr,
       MTOK, VTGT, D, D, D, VTGT, 0,0,0,0,0,0, 1,1, 1.f, 0);
}
